// PointNet2ClassificationSSG_64768106823698
// MI455X (gfx1250) — matrix operand order repaired and checked by reading, not compiled, not run on hardware
//
#include <hip/hip_runtime.h>

typedef __attribute__((ext_vector_type(16))) __bf16 v16bf;
typedef __attribute__((ext_vector_type(8)))  float  v8f;

#define BATCH 32
#define NPTS  4096
#define EPSV  1e-5f

// ---------------------------------------------------------------------------
// Generic WMMA layer: Y(M x 16cols) = relu(W * X + bias), bf16 in, f32 acc.
// actIn/actOut: LDS, column-major, column stride sIn/sOut (elems, mult of 32).
// wfrag: global bf16, pre-swizzled A-fragments: tile (kt,mt) at
//        ((kt*Mt+mt)*32 + lane)*16, 16 contiguous elems per lane.
// If gmax != nullptr: per-row max over this wave's 16 columns is atomically
// max-combined into gmax[row*gstride] (values >= 0 after ReLU, so unsigned
// bit-pattern atomicMax is exact).
// ---------------------------------------------------------------------------
__device__ __forceinline__ void layer_gemm(
    const __bf16* __restrict__ wfrag, const float* __restrict__ bias,
    int M, int Kt,
    const __bf16* actIn, int sIn,
    __bf16* actOut, int sOut,
    float* gmax, int gstride,
    int lane, int colBase)
{
  const int col  = colBase + (lane & 15);
  const int half = lane >> 4;
  const int Mt   = M >> 4;
  const __bf16* bcol = actIn + (size_t)col * sIn + half * 16;
  for (int mt = 0; mt < Mt; ++mt) {
    v8f acc = {};
    for (int kt = 0; kt < Kt; ++kt) {
      v16bf a  = *(const v16bf*)(wfrag + ((size_t)(kt * Mt + mt) * 32 + lane) * 16);
      v16bf bb = *(const v16bf*)(bcol + kt * 32);
      acc = __builtin_amdgcn_wmma_f32_16x16x32_bf16(false, a, false, bb,
                                                    (short)0, acc, false, false);
    }
#pragma unroll
    for (int i = 0; i < 8; ++i) {
      int row = mt * 16 + half * 8 + i;
      float v = acc[i] + bias[row];
      v = v > 0.f ? v : 0.f;
      if (actOut) actOut[(size_t)col * sOut + row] = (__bf16)v;
      if (gmax) {
        float m = v;
#pragma unroll
        for (int x = 1; x < 16; x <<= 1) {
          float o = __shfl_xor(m, x, 32);
          m = o > m ? o : m;
        }
        if ((lane & 15) == 0)
          atomicMax((unsigned int*)(gmax + (size_t)row * gstride), __float_as_uint(m));
      }
    }
  }
}

// ---------------------------------------------------------------------------
// Weight prep: fold BN scale, pad, swizzle into A-fragment layout, cvt bf16.
// One wave per 16x32 tile; lane l covers row mt*16+(l&15), K contiguous:
//   kb = (l<16)?0:16 ;  e -> kt*32+kb+e
// ---------------------------------------------------------------------------
__global__ void prep_w_kernel(const float* __restrict__ W,
                              const float* __restrict__ g,
                              const float* __restrict__ va,
                              int M, int K, int Mt, __bf16* __restrict__ out)
{
  int tile = blockIdx.x, lane = threadIdx.x;
  int kt = tile / Mt, mt = tile % Mt;
  int row = mt * 16 + (lane & 15);
  int kb  = (lane < 16) ? 0 : 16;
  float sc = 1.f;
  if (g && row < M) sc = g[row] * rsqrtf(va[row] + EPSV);
  __bf16* o = out + ((size_t)tile * 32 + lane) * 16;
#pragma unroll
  for (int e = 0; e < 16; ++e) {
    int kk = kt * 32 + kb + e;
    float v = (row < M && kk < K) ? W[(size_t)row * K + kk] * sc : 0.f;
    o[e] = (__bf16)v;
  }
}

__global__ void prep_b_kernel(const float* g, const float* be, const float* me,
                              const float* va, const float* bo,
                              int M, int Mp, float* out)
{
  int i = blockIdx.x * blockDim.x + threadIdx.x;
  if (i >= Mp) return;
  float v = 0.f;
  if (i < M) {
    if (g) {
      float s = g[i] * rsqrtf(va[i] + EPSV);
      v = be[i] + ((bo ? bo[i] : 0.f) - me[i]) * s;
    } else {
      v = bo ? bo[i] : 0.f;
    }
  }
  out[i] = v;
}

__global__ void zero_f32(float* p, int n)
{
  int i = blockIdx.x * blockDim.x + threadIdx.x;
  if (i < n) p[i] = 0.f;
}

// ---------------------------------------------------------------------------
// Farthest point sampling (matches reference: first index = 0, argmax picks
// first occurrence on ties). One block per batch; emits selected coords.
// ---------------------------------------------------------------------------
__global__ __launch_bounds__(256) void fps_kernel(
    const float* __restrict__ xyz, int stride, int n, int npoint,
    float* __restrict__ outXyz)
{
  __shared__ float dist[4096];
  __shared__ float rv[256];
  __shared__ int   ri[256];
  __shared__ int   s_last;
  int b = blockIdx.x, tid = threadIdx.x;
  const float* base = xyz + (size_t)b * n * stride;
  float* oX = outXyz + (size_t)b * npoint * 3;
  for (int j = tid; j < n; j += 256) dist[j] = 1e10f;
  if (tid == 0) { s_last = 0; oX[0] = base[0]; oX[1] = base[1]; oX[2] = base[2]; }
  __syncthreads();
  for (int i = 1; i < npoint; ++i) {
    int last = s_last;
    float px = base[(size_t)last * stride + 0];
    float py = base[(size_t)last * stride + 1];
    float pz = base[(size_t)last * stride + 2];
    float bv = -1.f; int bi = 0x7fffffff;
    for (int j = tid; j < n; j += 256) {
      const float* p = base + (size_t)j * stride;
      float dx = p[0] - px, dy = p[1] - py, dz = p[2] - pz;
      float d = dx * dx + dy * dy + dz * dz;
      float dd = dist[j];
      if (d < dd) { dist[j] = d; dd = d; }
      if (dd > bv) { bv = dd; bi = j; }
    }
    rv[tid] = bv; ri[tid] = bi;
    __syncthreads();
    for (int s2 = 128; s2 > 0; s2 >>= 1) {
      if (tid < s2) {
        if (rv[tid + s2] > rv[tid] ||
            (rv[tid + s2] == rv[tid] && ri[tid + s2] < ri[tid])) {
          rv[tid] = rv[tid + s2]; ri[tid] = ri[tid + s2];
        }
      }
      __syncthreads();
    }
    if (tid == 0) {
      int nx = ri[0]; s_last = nx;
      oX[(size_t)i * 3 + 0] = base[(size_t)nx * stride + 0];
      oX[(size_t)i * 3 + 1] = base[(size_t)nx * stride + 1];
      oX[(size_t)i * 3 + 2] = base[(size_t)nx * stride + 2];
    }
    __syncthreads();
  }
}

// ---------------------------------------------------------------------------
// SA1/SA2: one workgroup per (batch, group). Wave0 ball-query (first 64
// in-radius indices in increasing order, padded with first); 4 waves x 16
// columns run the 3-layer MLP in LDS; final layer max-pools via atomics.
// ---------------------------------------------------------------------------
__global__ __launch_bounds__(128) void sa_mlp_kernel(
    const float* __restrict__ pts, int n, int pstride,
    const float* __restrict__ feats, int Cfeat, int featFromPts,
    const float* __restrict__ centers, int S, float r2,
    const __bf16* __restrict__ w0, const float* __restrict__ b0, int M0, int Kt0,
    const __bf16* __restrict__ w1, const float* __restrict__ b1, int M1, int Kt1,
    const __bf16* __restrict__ w2, const float* __restrict__ b2, int M2, int Kt2,
    int capA,
    float* __restrict__ featOut)
{
  extern __shared__ char smem[];
  int*    gidx = (int*)smem;
  __bf16* bufA = (__bf16*)(smem + 256);
  __bf16* bufB = bufA + (size_t)64 * capA;
  int b = blockIdx.x / S, s = blockIdx.x % S;
  int tid = threadIdx.x, lane = tid & 31, wave = tid >> 5;
  const float* cen = centers + ((size_t)b * S + s) * 3;
  float cx = cen[0], cy = cen[1], cz = cen[2];
  const float* pbase = pts + (size_t)b * n * pstride;
  int K0p = Kt0 * 32;

  if (wave == 0) {
    int c = 0;
    for (int j0 = 0; j0 < n && c < 64; j0 += 32) {
      int j = j0 + lane;
      bool inr = false;
      if (j < n) {
        const float* p = pbase + (size_t)j * pstride;
        float dx = p[0] - cx, dy = p[1] - cy, dz = p[2] - cz;
        inr = (dx * dx + dy * dy + dz * dz) < r2;
      }
      unsigned m = (unsigned)__ballot(inr);
      int pos = c + __popc(m & ((1u << lane) - 1u));
      if (inr && pos < 64) gidx[pos] = j;
      c += __popc(m);
    }
    if (c > 64) c = 64;
    if (c == 0) { if (lane == 0) gidx[0] = 0; c = 1; }
    int g0 = gidx[0];
    for (int k2 = c + lane; k2 < 64; k2 += 32) gidx[k2] = g0;
  }
  __syncthreads();

  if (tid < 64) {
    int j = gidx[tid];
    const float* p = pbase + (size_t)j * pstride;
    __bf16* colp = bufA + (size_t)tid * K0p;
    colp[0] = (__bf16)(p[0] - cx);
    colp[1] = (__bf16)(p[1] - cy);
    colp[2] = (__bf16)(p[2] - cz);
    if (featFromPts) {
      for (int c2 = 0; c2 < Cfeat; ++c2) colp[3 + c2] = (__bf16)p[3 + c2];
    } else {
      const float* f = feats + (size_t)b * Cfeat * n + j;
      for (int c2 = 0; c2 < Cfeat; ++c2) colp[3 + c2] = (__bf16)f[(size_t)c2 * n];
    }
    for (int r = 3 + Cfeat; r < K0p; ++r) colp[r] = (__bf16)0.f;
  }
  __syncthreads();

  int colBase = wave * 16;
  layer_gemm(w0, b0, M0, Kt0, bufA, K0p, bufB, M0, nullptr, 0, lane, colBase);
  __syncthreads();
  layer_gemm(w1, b1, M1, Kt1, bufB, M0, bufA, M1, nullptr, 0, lane, colBase);
  __syncthreads();
  layer_gemm(w2, b2, M2, Kt2, bufA, M1, nullptr, 0,
             featOut + (size_t)b * M2 * S + s, S, lane, colBase);
}

// ---------------------------------------------------------------------------
// SA3: global grouping (absolute xyz + 256 features over 128 points).
// One workgroup per (batch, 32-point chunk), 2 waves.
// ---------------------------------------------------------------------------
__global__ __launch_bounds__(64) void sa3_kernel(
    const float* __restrict__ xyz2,   // [B][128][3]
    const float* __restrict__ feat2,  // [B][256][128]
    const __bf16* __restrict__ w0, const float* __restrict__ b0,
    const __bf16* __restrict__ w1, const float* __restrict__ b1,
    const __bf16* __restrict__ w2, const float* __restrict__ b2,
    float* __restrict__ code)         // [B][1024]
{
  extern __shared__ char smem[];
  __bf16* bufA = (__bf16*)smem;               // 32 cols * 512
  __bf16* bufB = bufA + (size_t)32 * 512;     // 32 cols * 1024
  int b = blockIdx.x >> 2, chunk = blockIdx.x & 3;
  int tid = threadIdx.x, lane = tid & 31, wave = tid >> 5;
  if (tid < 32) {
    int p = chunk * 32 + tid;
    __bf16* colp = bufA + (size_t)tid * 288;
    const float* x = xyz2 + ((size_t)b * 128 + p) * 3;
    colp[0] = (__bf16)x[0]; colp[1] = (__bf16)x[1]; colp[2] = (__bf16)x[2];
    const float* f = feat2 + (size_t)b * 256 * 128 + p;
    for (int c2 = 0; c2 < 256; ++c2) colp[3 + c2] = (__bf16)f[(size_t)c2 * 128];
    for (int r = 259; r < 288; ++r) colp[r] = (__bf16)0.f;
  }
  __syncthreads();
  int colBase = wave * 16;
  layer_gemm(w0, b0, 256,  9, bufA, 288, bufB, 256, nullptr, 0, lane, colBase);
  __syncthreads();
  layer_gemm(w1, b1, 512,  8, bufB, 256, bufA, 512, nullptr, 0, lane, colBase);
  __syncthreads();
  layer_gemm(w2, b2, 1024, 16, bufA, 512, nullptr, 0,
             code + (size_t)b * 1024, 1, lane, colBase);
}

// ---------------------------------------------------------------------------
// Folding head: 21 skeleton columns (padded to 32), 1026->256->256->128->3.
// One workgroup per batch, 2 waves.
// ---------------------------------------------------------------------------
__global__ __launch_bounds__(64) void fold_kernel(
    const float* __restrict__ skel,  // [2][21]
    const float* __restrict__ code,  // [B][1024]
    const __bf16* __restrict__ w0, const float* __restrict__ b0,
    const __bf16* __restrict__ w1, const float* __restrict__ b1,
    const __bf16* __restrict__ w2, const float* __restrict__ b2,
    const __bf16* __restrict__ wf, const float* __restrict__ bf_,
    float* __restrict__ out)         // [B][3][21]
{
  extern __shared__ char smem[];
  __bf16* bufA = (__bf16*)smem;                 // 32 cols * 1056
  __bf16* bufB = bufA + (size_t)32 * 1056;      // 32 cols * 256
  int b = blockIdx.x, tid = threadIdx.x, lane = tid & 31, wave = tid >> 5;
  if (tid < 32) {
    __bf16* colp = bufA + (size_t)tid * 1056;
    if (tid < 21) {
      colp[0] = (__bf16)skel[tid];
      colp[1] = (__bf16)skel[21 + tid];
      const float* cd = code + (size_t)b * 1024;
      for (int c2 = 0; c2 < 1024; ++c2) colp[2 + c2] = (__bf16)cd[c2];
      for (int r = 1026; r < 1056; ++r) colp[r] = (__bf16)0.f;
    } else {
      for (int r = 0; r < 1056; ++r) colp[r] = (__bf16)0.f;
    }
  }
  __syncthreads();
  int colBase = wave * 16;
  layer_gemm(w0, b0, 256, 33, bufA, 1056, bufB, 256, nullptr, 0, lane, colBase);
  __syncthreads();
  layer_gemm(w1, b1, 256,  8, bufB, 256, bufA, 256, nullptr, 0, lane, colBase);
  __syncthreads();
  layer_gemm(w2, b2, 128,  8, bufA, 256, bufB, 128, nullptr, 0, lane, colBase);
  __syncthreads();
  // final 3x128 linear (one 16x16 D tile; rows 0..2 valid, cols < 21)
  {
    int col = colBase + (lane & 15), half = lane >> 4;
    v8f acc = {};
    const __bf16* bp = bufB + (size_t)col * 128 + half * 16;
    for (int kt = 0; kt < 4; ++kt) {
      v16bf a  = *(const v16bf*)(wf + ((size_t)kt * 32 + lane) * 16);
      v16bf bb = *(const v16bf*)(bp + kt * 32);
      acc = __builtin_amdgcn_wmma_f32_16x16x32_bf16(false, a, false, bb,
                                                    (short)0, acc, false, false);
    }
    if (half == 0 && col < 21) {
#pragma unroll
      for (int i = 0; i < 3; ++i)
        out[((size_t)b * 3 + i) * 21 + col] = acc[i] + bf_[i];
    }
  }
}

// ---------------------------------------------------------------------------
// Host side
// ---------------------------------------------------------------------------
struct LayerIdx { int w, g, be, me, va, b; int M, K; };

// Flattening assumption: setup_inputs() dict in insertion order, params
// recursively (sa1[0..2], sa2[0..2], sa3[0..2], fold1[0..2], fold1_out),
// each layer dict in insertion order {W, gamma, beta, mean, var, [b]}.
static const LayerIdx LM[13] = {
  { 2,  3,  4,  5,  6, -1,   64,    6},
  { 7,  8,  9, 10, 11, -1,   64,   64},
  {12, 13, 14, 15, 16, -1,  128,   64},
  {17, 18, 19, 20, 21, -1,  128,  131},
  {22, 23, 24, 25, 26, -1,  128,  128},
  {27, 28, 29, 30, 31, -1,  256,  128},
  {32, 33, 34, 35, 36, -1,  256,  259},
  {37, 38, 39, 40, 41, -1,  512,  256},
  {42, 43, 44, 45, 46, -1, 1024,  512},
  {47, 48, 49, 50, 51, 52,  256, 1026},
  {53, 54, 55, 56, 57, 58,  256,  256},
  {59, 60, 61, 62, 63, 64,  128,  256},
  {65, -1, -1, -1, -1, 66,    3,  128},
};

extern "C" void kernel_launch(void* const* d_in, const int* in_sizes, int n_in,
                              void* d_out, int out_size, void* d_ws, size_t ws_size,
                              hipStream_t stream)
{
  (void)in_sizes; (void)n_in; (void)out_size; (void)ws_size;
  char* ws = (char*)d_ws;
  size_t cur = 0;
  auto alloc = [&](size_t bytes) -> char* {
    char* p = ws + cur;
    cur = (cur + bytes + 255) & ~(size_t)255;
    return p;
  };

  const float* pc   = (const float*)d_in[0];
  const float* skel = (const float*)d_in[1];

  __bf16* wfrag[13];
  float*  bias[13];

  // -- weight/bias prep (BN folding + bf16 fragment swizzle) --
  for (int l = 0; l < 13; ++l) {
    int M = LM[l].M, K = LM[l].K;
    int Mp = (M + 15) & ~15, Kp = (K + 31) & ~31;
    int Mt = Mp / 16, Kt = Kp / 32;
    wfrag[l] = (__bf16*)alloc((size_t)Mp * Kp * sizeof(__bf16));
    bias[l]  = (float*)alloc((size_t)Mp * sizeof(float));
    const float* W  = (const float*)d_in[LM[l].w];
    const float* g  = LM[l].g  >= 0 ? (const float*)d_in[LM[l].g]  : nullptr;
    const float* be = LM[l].be >= 0 ? (const float*)d_in[LM[l].be] : nullptr;
    const float* me = LM[l].me >= 0 ? (const float*)d_in[LM[l].me] : nullptr;
    const float* va = LM[l].va >= 0 ? (const float*)d_in[LM[l].va] : nullptr;
    const float* bo = LM[l].b  >= 0 ? (const float*)d_in[LM[l].b]  : nullptr;
    prep_w_kernel<<<Mt * Kt, 32, 0, stream>>>(W, g, va, M, K, Mt, wfrag[l]);
    prep_b_kernel<<<(Mp + 255) / 256, 256, 0, stream>>>(g, be, me, va, bo, M, Mp, bias[l]);
  }

  float* nxyz1 = (float*)alloc((size_t)BATCH * 512 * 3 * 4);
  float* feat1 = (float*)alloc((size_t)BATCH * 128 * 512 * 4);
  float* nxyz2 = (float*)alloc((size_t)BATCH * 128 * 3 * 4);
  float* feat2 = (float*)alloc((size_t)BATCH * 256 * 128 * 4);
  float* code  = (float*)alloc((size_t)BATCH * 1024 * 4);

  zero_f32<<<(BATCH * 128 * 512 + 255) / 256, 256, 0, stream>>>(feat1, BATCH * 128 * 512);
  zero_f32<<<(BATCH * 256 * 128 + 255) / 256, 256, 0, stream>>>(feat2, BATCH * 256 * 128);
  zero_f32<<<(BATCH * 1024 + 255) / 256, 256, 0, stream>>>(code, BATCH * 1024);

  // dynamic LDS budgets
  const int smem1 = 256 + 64 * 64  * 2 + 64 * 64  * 2;   // SA1: 16.6 KB
  const int smem2 = 256 + 64 * 160 * 2 + 64 * 128 * 2;   // SA2: 37 KB
  const int smem3 = 32 * 512 * 2 + 32 * 1024 * 2;        // SA3: 96 KB
  const int smem4 = 32 * 1056 * 2 + 32 * 256 * 2;        // fold: 82 KB
  (void)hipFuncSetAttribute((const void*)sa_mlp_kernel,
      hipFuncAttributeMaxDynamicSharedMemorySize, smem2);
  (void)hipFuncSetAttribute((const void*)sa3_kernel,
      hipFuncAttributeMaxDynamicSharedMemorySize, smem3);
  (void)hipFuncSetAttribute((const void*)fold_kernel,
      hipFuncAttributeMaxDynamicSharedMemorySize, smem4);

  // SA1
  fps_kernel<<<BATCH, 256, 0, stream>>>(pc, 6, NPTS, 512, nxyz1);
  sa_mlp_kernel<<<BATCH * 512, 128, smem1, stream>>>(
      pc, NPTS, 6, nullptr, 3, 1, nxyz1, 512, 0.01f,
      wfrag[0], bias[0], 64, 1,
      wfrag[1], bias[1], 64, 2,
      wfrag[2], bias[2], 128, 2,
      /*capA=*/64, feat1);

  // SA2
  fps_kernel<<<BATCH, 256, 0, stream>>>(nxyz1, 3, 512, 128, nxyz2);
  sa_mlp_kernel<<<BATCH * 128, 128, smem2, stream>>>(
      nxyz1, 512, 3, feat1, 128, 0, nxyz2, 128, 0.04f,
      wfrag[3], bias[3], 128, 5,
      wfrag[4], bias[4], 128, 4,
      wfrag[5], bias[5], 256, 4,
      /*capA=*/160, feat2);

  // SA3 (global)
  sa3_kernel<<<BATCH * 4, 64, smem3, stream>>>(
      nxyz2, feat2,
      wfrag[6], bias[6], wfrag[7], bias[7], wfrag[8], bias[8], code);

  // Folding head
  fold_kernel<<<BATCH, 64, smem4, stream>>>(
      skel, code,
      wfrag[9], bias[9], wfrag[10], bias[10], wfrag[11], bias[11],
      wfrag[12], bias[12], (float*)d_out);
}
